// AttentionRouting_42056319762807
// MI455X (gfx1250) — matrix operand order repaired and checked by reading, not compiled, not run on hardware
//
#include <hip/hip_runtime.h>
#include <hip/hip_bf16.h>

typedef __attribute__((ext_vector_type(16))) _Float16 v16h;
typedef __attribute__((ext_vector_type(8)))  float    v8f;
typedef __attribute__((ext_vector_type(4)))  float    f32x4;

// Problem constants (match reference)
static constexpr int NB   = 32;    // batch
static constexpr int NG   = 8;     // groups
static constexpr int CIN  = 64;    // channels per group
static constexpr int HID  = 512;   // CIN * RATIO
static constexpr int COUT = 256;   // CAPS * OUT
static constexpr int HW   = 4096;  // 64*64 spatial

// ---------------------------------------------------------------------------
// Kernel 1: global average pool over spatial. embedding [B, G*CIN, 64, 64]
// -> pooled [B*G*CIN]. One block per (b, channel) row; NT loads (read-once).
// ---------------------------------------------------------------------------
__global__ void __launch_bounds__(256)
pool_kernel(const float* __restrict__ emb, float* __restrict__ pooled) {
    __shared__ float red[256];
    const int row = blockIdx.x;                       // b*512 + c
    const f32x4* p = (const f32x4*)(emb + (size_t)row * HW);
    const int t = threadIdx.x;
    float s = 0.f;
#pragma unroll
    for (int j = 0; j < 4; ++j) {
        f32x4 v = __builtin_nontemporal_load(&p[t + 256 * j]);
        s += v.x + v.y + v.z + v.w;
    }
    red[t] = s;
    __syncthreads();
    for (int off = 128; off > 0; off >>= 1) {
        if (t < off) red[t] += red[t + off];
        __syncthreads();
    }
    if (t == 0) pooled[row] = red[0] * (1.f / (float)HW);
}

// ---------------------------------------------------------------------------
// Kernel 2: per-group 2-layer MLP via WMMA.
//   h    = relu(pooled[:,g,:] @ w1[g]^T + b1[g])   (M=32, K=64,  N=512)
//   atts =      h             @ w2[g]^T + b2[g]    (M=32, K=512, N=256)
// One workgroup per group g; 8 waves. f16 operands, f32 accumulate.
// ---------------------------------------------------------------------------
__device__ __forceinline__ int a_krel(int lh, int j) {
    // 16-bit A 16x32 fragment: relative K held in f16 slot j of lane half lh.
    // lanes 0-15 (lh=0): K = {0..15}; lanes 16-31 (lh=1): K = {16..31}
    return lh * 16 + j;
}

__global__ void __launch_bounds__(256)
mlp_kernel(const float* __restrict__ pooled,
           const float* __restrict__ w1, const float* __restrict__ b1,
           const float* __restrict__ w2, const float* __restrict__ b2,
           float* __restrict__ atts) {
    const int g    = blockIdx.x;
    const int t    = threadIdx.x;
    const int lane = t & 31;
    const int wave = t >> 5;
    const int ln   = lane & 15;   // N index within tile / M index within tile
    const int lh   = lane >> 4;   // lane half

    __shared__ float    sPooled[NB * CIN];   // 8 KB  (A source, GEMM1)
    __shared__ _Float16 sH[NB * HID];        // 32 KB (h in f16 = A source, GEMM2)

    // Stage pooled[:, g, :] into LDS
    for (int i = t; i < NB * CIN; i += 256) {
        int b = i >> 6, k = i & 63;
        sPooled[i] = pooled[((size_t)b * NG + g) * CIN + k];
    }
    __syncthreads();

    // ---------------- GEMM1: 32x64 @ 64x512, 32 N-tiles over 8 waves -------
    for (int tt = 0; tt < 4; ++tt) {
        const int n0 = (wave * 4 + tt) * 16;
        const int n  = n0 + ln;
        const float* w1row = w1 + ((size_t)g * HID + n) * CIN + lh * 16;
#pragma unroll
        for (int mt = 0; mt < 2; ++mt) {
            v8f acc = {};
#pragma unroll
            for (int kb = 0; kb < CIN; kb += 32) {
                v16h a, bf;
#pragma unroll
                for (int j = 0; j < 16; ++j) {
                    a[j]  = (_Float16)sPooled[(mt * 16 + ln) * CIN + kb + a_krel(lh, j)];
                    bf[j] = (_Float16)w1row[kb + j];   // B(k,n): lane ln = n, K = kb + lh*16 + j
                }
                acc = __builtin_amdgcn_wmma_f32_16x16x32_f16(
                    false, a, false, bf, (short)0, acc, false, false);
            }
            const float bias = b1[(size_t)g * HID + n];
#pragma unroll
            for (int r = 0; r < 8; ++r) {
                int m = mt * 16 + lh * 8 + r;          // C/D layout: M = r + 8*lh
                float hv = acc[r] + bias;
                hv = fmaxf(hv, 0.f);
                sH[m * HID + n] = (_Float16)hv;
            }
        }
    }
    __syncthreads();

    // ---------------- GEMM2: 32x512 @ 512x256, 16 N-tiles over 8 waves -----
    for (int tt = 0; tt < 2; ++tt) {
        const int n0 = (wave * 2 + tt) * 16;
        const int n  = n0 + ln;
        const float* w2row = w2 + ((size_t)g * COUT + n) * HID + lh * 16;
#pragma unroll
        for (int mt = 0; mt < 2; ++mt) {
            v8f acc = {};
            for (int kb = 0; kb < HID; kb += 32) {
                v16h a, bf;
#pragma unroll
                for (int j = 0; j < 16; ++j) {
                    a[j]  = sH[(mt * 16 + ln) * HID + kb + a_krel(lh, j)];
                    bf[j] = (_Float16)w2row[kb + j];
                }
                acc = __builtin_amdgcn_wmma_f32_16x16x32_f16(
                    false, a, false, bf, (short)0, acc, false, false);
            }
            const float bias = b2[(size_t)g * COUT + n];
#pragma unroll
            for (int r = 0; r < 8; ++r) {
                int b = mt * 16 + lh * 8 + r;          // M = batch
                atts[((size_t)b * NG + g) * COUT + n] = acc[r] + bias;
            }
        }
    }
}

// ---------------------------------------------------------------------------
// Kernel 3: dynamic routing (3 iters) + sigmoid -> gate [B, 256].
// One block per batch; thread t = (caps c = t/64, channel ch = t%64).
// ---------------------------------------------------------------------------
__global__ void __launch_bounds__(256)
route_kernel(const float* __restrict__ atts, float* __restrict__ gate) {
    const int b  = blockIdx.x;
    const int t  = threadIdx.x;
    const int c  = t >> 6;
    const int ch = t & 63;

    __shared__ float beta[32];     // [g*4 + c]
    __shared__ float red[2048];    // reduction scratch [g][256]
    __shared__ float nrmv[4];

    float xr[8];
#pragma unroll
    for (int g = 0; g < 8; ++g)
        xr[g] = atts[((size_t)b * NG + g) * COUT + c * 64 + ch];

    if (t < 32) beta[t] = 0.f;
    __syncthreads();

    float routed = 0.f;
    for (int it = 0; it < 3; ++it) {
        // softmax over capsules (axis=2) per group, evaluated at my c
        float v = 0.f;
#pragma unroll
        for (int g = 0; g < 8; ++g) {
            float mx = beta[g * 4 + 0];
            for (int cc = 1; cc < 4; ++cc) mx = fmaxf(mx, beta[g * 4 + cc]);
            float den = 0.f;
            for (int cc = 0; cc < 4; ++cc) den += __expf(beta[g * 4 + cc] - mx);
            float alpha = __expf(beta[g * 4 + c] - mx) / den;
            v += alpha * xr[g];                     // sum over groups
        }
        if (it == 2) { routed = v; break; }

        // L2 norm over channels per capsule
        red[t] = v * v;
        __syncthreads();
        if (ch == 0) {
            float s = 0.f;
            for (int k = 0; k < 64; ++k) s += red[c * 64 + k];
            nrmv[c] = fmaxf(sqrtf(s), 1e-12f);
        }
        __syncthreads();
        const float vn = v / nrmv[c];

        // beta[g][c] += sum_ch vn * xr[g]
#pragma unroll
        for (int g = 0; g < 8; ++g) red[g * 256 + t] = vn * xr[g];
        __syncthreads();
        if (t < 32) {
            int g = t >> 2, cc = t & 3;
            float s = 0.f;
            for (int k = 0; k < 64; ++k) s += red[g * 256 + cc * 64 + k];
            beta[g * 4 + cc] += s;
        }
        __syncthreads();
    }
    gate[(size_t)b * COUT + c * 64 + ch] = 1.f / (1.f + __expf(-routed));
}

// ---------------------------------------------------------------------------
// Kernel 4: out = sigmoid-gate (per b,channel) * x. 256 MB streamed, NT.
// One block per (b, channel) row of 4096 pixels.
// ---------------------------------------------------------------------------
__global__ void __launch_bounds__(256)
gate_mul_kernel(const float* __restrict__ x, const float* __restrict__ gate,
                float* __restrict__ out) {
    const int row = blockIdx.x;                      // b*256 + C
    const float s = gate[row];
    const f32x4* px = (const f32x4*)(x + (size_t)row * HW);
    f32x4* po       = (f32x4*)(out + (size_t)row * HW);
    const int t = threadIdx.x;
#pragma unroll
    for (int j = 0; j < 4; ++j) {
        f32x4 v = __builtin_nontemporal_load(&px[t + 256 * j]);
        f32x4 r = { s * v.x, s * v.y, s * v.z, s * v.w };
        __builtin_nontemporal_store(r, &po[t + 256 * j]);
    }
}

// ---------------------------------------------------------------------------
extern "C" void kernel_launch(void* const* d_in, const int* in_sizes, int n_in,
                              void* d_out, int out_size, void* d_ws, size_t ws_size,
                              hipStream_t stream) {
    (void)in_sizes; (void)n_in; (void)out_size; (void)ws_size;
    const float* emb = (const float*)d_in[0];
    const float* x   = (const float*)d_in[1];
    const float* w1  = (const float*)d_in[2];
    const float* b1  = (const float*)d_in[3];
    const float* w2  = (const float*)d_in[4];
    const float* b2  = (const float*)d_in[5];
    float* out = (float*)d_out;

    float* pooled = (float*)d_ws;                  // 32*512      = 16384 f32
    float* atts   = pooled + NB * NG * CIN;        // 32*8*256    = 65536 f32
    float* gate   = atts   + NB * NG * COUT;       // 32*256      =  8192 f32

    pool_kernel    <<<NB * NG * CIN, 256, 0, stream>>>(emb, pooled);
    mlp_kernel     <<<NG,            256, 0, stream>>>(pooled, w1, b1, w2, b2, atts);
    route_kernel   <<<NB,            256, 0, stream>>>(atts, gate);
    gate_mul_kernel<<<NB * COUT,     256, 0, stream>>>(x, gate, out);
}
